// IntNetwork_16870631539081
// MI455X (gfx1250) — compile-verified, matrix operand order checked by reading, not run on hardware
//
#include <hip/hip_runtime.h>
#include <hip/hip_bf16.h>

// ---------------------------------------------------------------------------
// Shapes: B=8, OBJ(N)=128, INP=64, HID(H)=256, D=3*INP=192
// out = decoder(mean_j MLP3(relu(ea[b,i] + eb[b,j] + b_enc)))
// ---------------------------------------------------------------------------

typedef __bf16 bf16_t;
typedef __attribute__((ext_vector_type(16))) __bf16 v16bf;
typedef __attribute__((ext_vector_type(8)))  float  v8f;
typedef unsigned int u32x4 __attribute__((ext_vector_type(4)));

union Frag {            // one WMMA 16x16x32 bf16 operand: 16 bf16 per lane = 32 B
    v16bf v;
    u32x4 q[2];
};

#define NB    128       // objects per batch
#define HID_  256
#define DIN   192

// ---------------------------------------------------------------------------
// Kernel 0: convert w1/w2/w3 (f32, [k_in=256][n_out=256]) -> bf16 transposed
// [n_out][k_in], 3 layers contiguous in workspace.
// ---------------------------------------------------------------------------
__global__ __launch_bounds__(256) void wprep_kernel(
    const float* __restrict__ w1, const float* __restrict__ w2,
    const float* __restrict__ w3, bf16_t* __restrict__ wT)
{
    int idx   = blockIdx.x * 256 + threadIdx.x;   // 0 .. 3*65536-1
    int layer = idx >> 16;
    int o     = idx & 65535;
    int n     = o >> 8;        // output channel (row of wT)
    int k     = o & 255;       // input channel
    const float* w = (layer == 0) ? w1 : ((layer == 1) ? w2 : w3);
    wT[idx] = (bf16_t)w[k * HID_ + n];
}

// ---------------------------------------------------------------------------
// Kernel 1: encoder.  ea[r][h] = sum_d x[r][d] * w_enc[d][h]
//           eb[r][h] = sum_d x[r][d] * w_enc[192+d][h],  r = b*128+n
// x is the concat of x0|x1|x2 along d. Tiny GEMM -> plain VALU.
// ---------------------------------------------------------------------------
__global__ __launch_bounds__(256) void encoder_kernel(
    const float* __restrict__ x0, const float* __restrict__ x1,
    const float* __restrict__ x2, const float* __restrict__ w_enc,
    float* __restrict__ ea, float* __restrict__ eb)
{
    __shared__ float xr[DIN];
    const int r   = blockIdx.x;          // 0..1023
    const int tid = threadIdx.x;         // 0..255 -> output column h
    if (tid < DIN) {
        int d = tid;
        float v = (d < 64) ? x0[r * 64 + d]
                : (d < 128) ? x1[r * 64 + (d - 64)]
                            : x2[r * 64 + (d - 128)];
        xr[d] = v;
    }
    __syncthreads();
    float accA = 0.f, accB = 0.f;
    for (int d = 0; d < DIN; ++d) {
        float xv = xr[d];
        accA += xv * w_enc[d * HID_ + tid];
        accB += xv * w_enc[(d + DIN) * HID_ + tid];
    }
    ea[r * HID_ + tid] = accA;
    eb[r * HID_ + tid] = accB;
}

// ---------------------------------------------------------------------------
// Kernel 2: fused pairwise MLP + mean pool + decoder. One WG per (b, i).
// LDS: wLDS 128KB | actA 64KB | actB 64KB | eaRow/benc/pooled/hd1 4KB |
//      biasL 3KB (b1,b2,b3 staged so the GEMM epilogue never touches VMEM).
// 8 waves; wave w owns M-tile rows [16w, 16w+16); loops 16 N-tiles x 8 K-steps
// of v_wmma_f32_16x16x32_bf16 per layer.
// ---------------------------------------------------------------------------
__global__ __launch_bounds__(256) void relnet_main_kernel(
    const float*  __restrict__ ea,   const float* __restrict__ eb,
    const float*  __restrict__ benc,
    const bf16_t* __restrict__ w1T,  const bf16_t* __restrict__ w2T,
    const bf16_t* __restrict__ w3T,
    const float*  __restrict__ b1,   const float* __restrict__ b2,
    const float*  __restrict__ b3,
    const float*  __restrict__ wd1,  const float* __restrict__ bd1,
    const float*  __restrict__ wd2,  const float* __restrict__ bd2,
    float* __restrict__ out)
{
    extern __shared__ char smem[];
    bf16_t* wLDS   = (bf16_t*)(smem);                      // 256*256 bf16 = 131072 B
    bf16_t* actA   = (bf16_t*)(smem + 131072);             // 128*256 bf16 =  65536 B
    bf16_t* actB   = (bf16_t*)(smem + 131072 + 65536);     //                 65536 B
    float*  eaRow  = (float*)(smem + 262144);              // 256 f32
    float*  bencL  = eaRow + HID_;                         // 256 f32
    float*  pooled = bencL + HID_;                         // 256 f32
    float*  hd1    = pooled + HID_;                        // 256 f32
    float*  biasL  = hd1 + HID_;                           // 3*256 f32

    const int tid = threadIdx.x;
    const int blk = blockIdx.x;          // 0..1023
    const int b   = blk >> 7;
    const int i   = blk & 127;

    // Stage ea row i, encoder bias, layer biases; zero pooled accumulator.
    eaRow[tid]  = ea[((b << 7) + i) * HID_ + tid];
    bencL[tid]  = benc[tid];
    pooled[tid] = 0.f;
    biasL[tid]            = b1[tid];
    biasL[HID_ + tid]     = b2[tid];
    biasL[2 * HID_ + tid] = b3[tid];
    __syncthreads();

    // Build h0[j][k] = relu(ea_i[k] + eb[b,j][k] + benc[k]) as bf16 in actA.
    for (int idx = tid; idx < NB * HID_; idx += 256) {
        int j = idx >> 8, k = idx & 255;
        float v = eaRow[k] + eb[((b << 7) + j) * HID_ + k] + bencL[k];
        actA[idx] = (bf16_t)fmaxf(v, 0.f);
    }

    const int lane   = tid & 31;
    const int wave   = tid >> 5;      // 0..7 -> M-tile
    const int hi     = lane >> 4;     // K-chunk selector per ISA A/B layout
    const int lm     = lane & 15;     // row (A) / column (B) within tile
    const int m_base = wave << 4;

    const bf16_t* wTs[3] = {w1T, w2T, w3T};

#pragma unroll
    for (int layer = 0; layer < 3; ++layer) {
        __syncthreads();   // all waves done with previous layer's wLDS / dst
        // Cooperative 128 KB weight slab load (bf16, pre-transposed [n][k]).
        {
            const u32x4* gw = (const u32x4*)wTs[layer];
            u32x4*       lw = (u32x4*)wLDS;
            for (int idx = tid; idx < (HID_ * HID_ * 2) / 16; idx += 256)
                lw[idx] = gw[idx];
        }
        if (layer < 2) __builtin_prefetch(wTs[layer + 1], 0, 1);  // warm next layer (L2)
        __syncthreads();

        const bf16_t* src = (layer == 1) ? actB : actA;
        bf16_t*       dst = (layer == 1) ? actA : actB;

        // Hoist all 8 A fragments for this wave's 16-row M-tile.
        // Lane layout (16-bit A 16x32): lanes 0-15: K 0-7 & 16-23;
        //                               lanes 16-31: K 8-15 & 24-31.
        Frag afr[8];
        {
            const bf16_t* arow = src + (m_base + lm) * HID_;
#pragma unroll
            for (int kk = 0; kk < 8; ++kk) {
                int k0 = kk * 32 + hi * 8;
                afr[kk].q[0] = *(const u32x4*)(arow + k0);
                afr[kk].q[1] = *(const u32x4*)(arow + k0 + 16);
            }
        }

        for (int nt = 0; nt < 16; ++nt) {
            const int n_base = nt << 4;
            v8f acc = {};
            const bf16_t* brow = wLDS + (n_base + lm) * HID_;  // wT row n = B column n
#pragma unroll
            for (int kk = 0; kk < 8; ++kk) {
                Frag bfr;
                int k0 = kk * 32 + hi * 8;
                bfr.q[0] = *(const u32x4*)(brow + k0);
                bfr.q[1] = *(const u32x4*)(brow + k0 + 16);
                acc = __builtin_amdgcn_wmma_f32_16x16x32_bf16(
                    false, afr[kk].v, false, bfr.v, (short)0, acc, false, false);
            }
            const int nn = n_base + lm;
            if (layer < 2) {
                // D tile layout: lane (hi,lm) col nn, VGPR v -> row m_base+8*hi+v
                float bias = biasL[layer * HID_ + nn];   // LDS, no VMEM in hot loop
#pragma unroll
                for (int v = 0; v < 8; ++v) {
                    float x = acc[v] + bias;
                    dst[(m_base + hi * 8 + v) * HID_ + nn] = (bf16_t)fmaxf(x, 0.f);
                }
            } else {
                // Layer 3: no relu; reduce this tile's 8 rows, pool over j via LDS atomics.
                float s = acc[0] + acc[1] + acc[2] + acc[3]
                        + acc[4] + acc[5] + acc[6] + acc[7];
                atomicAdd(&pooled[nn], s);
            }
        }
    }
    __syncthreads();

    // pooled = colsum/128 + b3 (bias commutes with the mean over j)
    float pv = pooled[tid] * (1.0f / 128.0f) + biasL[2 * HID_ + tid];
    pooled[tid] = pv;
    __syncthreads();

    // Decoder mat-vec 1: hd1 = relu(pooled @ wd1 + bd1), wd1 is [256][256]
    {
        float a = bd1[tid];
        for (int k = 0; k < HID_; ++k)
            a += pooled[k] * wd1[k * HID_ + tid];
        hd1[tid] = fmaxf(a, 0.f);
    }
    __syncthreads();

    // Decoder mat-vec 2: out = hd1 @ wd2 + bd2, wd2 is [256][64]
    if (tid < 64) {
        float o = bd2[tid];
        for (int k = 0; k < HID_; ++k)
            o += hd1[k] * wd2[k * 64 + tid];
        out[blk * 64 + tid] = o;
    }
}

// ---------------------------------------------------------------------------
// Launcher. Inputs (setup_inputs order):
//  0:x0 1:x1 2:x2 3:w_enc 4:b_enc 5:w1 6:b1 7:w2 8:b2 9:w3 10:b3
//  11:wd1 12:bd1 13:wd2 14:bd2
// Workspace layout: [wT bf16 3*65536] [ea f32 1024*256] [eb f32 1024*256]
// ---------------------------------------------------------------------------
extern "C" void kernel_launch(void* const* d_in, const int* in_sizes, int n_in,
                              void* d_out, int out_size, void* d_ws, size_t ws_size,
                              hipStream_t stream) {
    const float* x0    = (const float*)d_in[0];
    const float* x1    = (const float*)d_in[1];
    const float* x2    = (const float*)d_in[2];
    const float* w_enc = (const float*)d_in[3];
    const float* b_enc = (const float*)d_in[4];
    const float* w1    = (const float*)d_in[5];
    const float* b1    = (const float*)d_in[6];
    const float* w2    = (const float*)d_in[7];
    const float* b2    = (const float*)d_in[8];
    const float* w3    = (const float*)d_in[9];
    const float* b3    = (const float*)d_in[10];
    const float* wd1   = (const float*)d_in[11];
    const float* bd1   = (const float*)d_in[12];
    const float* wd2   = (const float*)d_in[13];
    const float* bd2   = (const float*)d_in[14];

    char* ws = (char*)d_ws;
    bf16_t* wT = (bf16_t*)ws;                              // 393,216 B
    float*  ea = (float*)(ws + 393216);                    // 1,048,576 B
    float*  eb = (float*)(ws + 393216 + 1048576);          // 1,048,576 B

    wprep_kernel<<<768, 256, 0, stream>>>(w1, w2, w3, wT);
    encoder_kernel<<<1024, 256, 0, stream>>>(x0, x1, x2, w_enc, ea, eb);

    // 128K (weights) + 2*64K (bf16 activations) + 4KB vectors + 3KB biases
    size_t smem = 131072 + 65536 + 65536 + 4 * HID_ * sizeof(float)
                + 3 * HID_ * sizeof(float);                // 269,312 B < 320 KB
    relnet_main_kernel<<<1024, 256, smem, stream>>>(
        ea, eb, b_enc,
        wT, wT + 65536, wT + 131072,
        b1, b2, b3, wd1, bd1, wd2, bd2,
        (float*)d_out);
}
